// Attention_17970143166479
// MI455X (gfx1250) — matrix operand order repaired and checked by reading, not compiled, not run on hardware
//
#include <hip/hip_runtime.h>
#include <hip/hip_bf16.h>

typedef __bf16 bf16_t;
typedef __attribute__((ext_vector_type(16))) __bf16 bf16x16;
typedef __attribute__((ext_vector_type(8)))  float  f32x8;

union FragAB { bf16x16 v; uint4 u[2]; };

#define KC 32
#define BM 128
#define BN 128
#define NT 256

// ---------------------------------------------------------------------------
// f32 -> bf16 elementwise convert (grid-stride)
// ---------------------------------------------------------------------------
__global__ void cvt_f32_bf16(const float* __restrict__ in, bf16_t* __restrict__ out,
                             long long n) {
  long long i = (long long)blockIdx.x * blockDim.x + threadIdx.x;
  long long stride = (long long)gridDim.x * blockDim.x;
  for (; i < n; i += stride) out[i] = (bf16_t)in[i];
}

// ---------------------------------------------------------------------------
// Per-batch 1024x1024 transpose with f32 -> bf16 convert: out[h][s] = in[s][h].
// 32x32 LDS tile (+1 pad), both global phases fully coalesced.
// ---------------------------------------------------------------------------
__global__ __launch_bounds__(256) void transpose_cvt_bf16(
    const float* __restrict__ in,   // [B][1024][1024] (s, h)
    bf16_t* __restrict__ out)       // [B][1024][1024] (h, s)
{
  __shared__ float tile[32][33];
  const int tx = threadIdx.x, ty = threadIdx.y;      // block (32, 8)
  const int s0 = blockIdx.x * 32, h0 = blockIdx.y * 32;
  const size_t zoff = (size_t)blockIdx.z * 1024 * 1024;
  const float* ip = in + zoff;
  bf16_t* op = out + zoff;
#pragma unroll
  for (int r = 0; r < 4; ++r)
    tile[ty + r * 8][tx] = ip[(size_t)(s0 + ty + r * 8) * 1024 + h0 + tx];
  __syncthreads();
#pragma unroll
  for (int r = 0; r < 4; ++r)
    op[(size_t)(h0 + ty + r * 8) * 1024 + s0 + tx] = (bf16_t)tile[tx][ty + r * 8];
}

// ---------------------------------------------------------------------------
// Generic WMMA GEMM:  C[z] = scale * (A[z] x B[z]^T) + bias
//   A: M x K row-major (f32 or bf16; converted to bf16 during LDS staging)
//   B: N x K row-major bf16 (row n = logical B column n)
//   C: M x N row-major, f32 or bf16
// Block tile 128x128, K-chunk 32, 8 wave32s: wave (wm 0..3, wn 0..1) computes a
// 32x64 output patch = 2 A-frags x 4 B-frags = 8 WMMAs per K-chunk.
// Double-buffered LDS; global prefetch of chunk k+1 overlaps WMMA of chunk k.
// ---------------------------------------------------------------------------
template<bool A_F32, bool HAS_BIAS, bool OUT_BF16>
__global__ __launch_bounds__(NT) void gemm_wmma(
    const void* __restrict__ Av, const bf16_t* __restrict__ Bm,
    const float* __restrict__ bias, void* __restrict__ Cv,
    int M, int N, int K, int lda, int ldb, int ldc,
    long long sA, long long sB, long long sC, float scale)
{
  __shared__ __attribute__((aligned(16))) bf16_t As[2][BM * KC];
  __shared__ __attribute__((aligned(16))) bf16_t Bs[2][BN * KC];

  const int tid  = threadIdx.x;
  const int lane = tid & 31;
  const int wv   = tid >> 5;   // 0..7
  const int wm   = wv & 3;     // 0..3 -> 32-row patch within BM
  const int wn   = wv >> 2;    // 0..1 -> 64-col half within BN
  const int half = lane >> 4;  // 0/1
  const int l16  = lane & 15;

  const int bm0 = blockIdx.y * BM;
  const int bn0 = blockIdx.x * BN;
  const int z   = blockIdx.z;

  const float*  Af = (const float*)Av  + (size_t)z * (A_F32 ? sA : 0);
  const bf16_t* Ab = (const bf16_t*)Av + (size_t)z * (A_F32 ? 0 : sA);
  const bf16_t* Bp = Bm + (size_t)z * sB;

  // ---- per-thread staging descriptors (fixed, fully unrolled) ----
  const float4* pAf[4]; int oAf[4];   // A f32: 1024 float4/tile -> 4 seg/thread
  const uint4*  pAh[2]; int oAh[2];   // A bf16: 512 uint4/tile -> 2 seg/thread
  const uint4*  pBh[2]; int oBh[2];   // B bf16: 512 uint4/tile -> 2 seg/thread

  if constexpr (A_F32) {
#pragma unroll
    for (int s = 0; s < 4; ++s) {
      const int i = tid + s * NT, m = i >> 3, k4 = (i & 7) << 2;   // 8 float4/row
      pAf[s] = (const float4*)(Af + (size_t)(bm0 + m) * lda + k4);
      oAf[s] = m * KC + k4;
    }
  } else {
#pragma unroll
    for (int s = 0; s < 2; ++s) {
      const int i = tid + s * NT, m = i >> 2, k8 = (i & 3) << 3;   // 4 uint4/row
      pAh[s] = (const uint4*)(Ab + (size_t)(bm0 + m) * lda + k8);
      oAh[s] = m * KC + k8;
    }
  }
#pragma unroll
  for (int s = 0; s < 2; ++s) {
    const int i = tid + s * NT, n = i >> 2, k8 = (i & 3) << 3;
    pBh[s] = (const uint4*)(Bp + (size_t)(bn0 + n) * ldb + k8);
    oBh[s] = n * KC + k8;
  }

  // prefetch registers
  float4 rAf[4]; uint4 rAh[2]; uint4 rBh[2];

  auto prefetch = [&]() {
    if constexpr (A_F32) {
#pragma unroll
      for (int s = 0; s < 4; ++s) { rAf[s] = *pAf[s]; pAf[s] += KC / 4; }
    } else {
#pragma unroll
      for (int s = 0; s < 2; ++s) { rAh[s] = *pAh[s]; pAh[s] += KC / 8; }
    }
#pragma unroll
    for (int s = 0; s < 2; ++s) { rBh[s] = *pBh[s]; pBh[s] += KC / 8; }
  };

  auto commit = [&](int b) {
    if constexpr (A_F32) {
#pragma unroll
      for (int s = 0; s < 4; ++s) {
        bf16_t* p = &As[b][oAf[s]];
        p[0] = (bf16_t)rAf[s].x; p[1] = (bf16_t)rAf[s].y;
        p[2] = (bf16_t)rAf[s].z; p[3] = (bf16_t)rAf[s].w;
      }
    } else {
#pragma unroll
      for (int s = 0; s < 2; ++s) *(uint4*)&As[b][oAh[s]] = rAh[s];
    }
#pragma unroll
    for (int s = 0; s < 2; ++s) *(uint4*)&Bs[b][oBh[s]] = rBh[s];
  };

  f32x8 acc[2][4];
#pragma unroll
  for (int r = 0; r < 2; ++r)
#pragma unroll
    for (int j = 0; j < 4; ++j)
#pragma unroll
      for (int e = 0; e < 8; ++e) acc[r][j][e] = 0.0f;

  // fill buffer 0
  prefetch();
  commit(0);

  const int nk = K / KC;
  int buf = 0;
  for (int kc = 0; kc < nk; ++kc) {
    __syncthreads();
    const bool more = (kc + 1 < nk);
    if (more) prefetch();     // global loads for chunk kc+1 overlap compute

    // ---- load all fragments, then 8 back-to-back WMMAs ----
    const bf16_t* as = As[buf];
    const bf16_t* bs = Bs[buf];
    FragAB fa[2], fb[4];
#pragma unroll
    for (int r = 0; r < 2; ++r) {
      const bf16_t* pa = as + (wm * 32 + r * 16 + l16) * KC;
      fa[r].u[0] = *(const uint4*)(pa + half * 8);        // K = half*8 .. +7
      fa[r].u[1] = *(const uint4*)(pa + 16 + half * 8);   // K = 16+half*8 .. +7
    }
#pragma unroll
    for (int j = 0; j < 4; ++j) {
      const bf16_t* pb = bs + (wn * 64 + j * 16 + l16) * KC;
      fb[j].u[0] = *(const uint4*)(pb + half * 8);
      fb[j].u[1] = *(const uint4*)(pb + 16 + half * 8);
    }
#pragma unroll
    for (int r = 0; r < 2; ++r)
#pragma unroll
      for (int j = 0; j < 4; ++j)
        acc[r][j] = __builtin_amdgcn_wmma_f32_16x16x32_bf16(
            false, fa[r].v, false, fb[j].v, (short)0, acc[r][j], false, false);

    if (more) commit(buf ^ 1);
    buf ^= 1;
  }

  // ---- epilogue: C/D layout — lane l: col = l&15, rows (l>>4)*8 + v ----
#pragma unroll
  for (int r = 0; r < 2; ++r) {
#pragma unroll
    for (int j = 0; j < 4; ++j) {
      const int col = bn0 + wn * 64 + j * 16 + l16;
      const float bv = HAS_BIAS ? bias[col] : 0.0f;
#pragma unroll
      for (int v = 0; v < 8; ++v) {
        const int row = bm0 + wm * 32 + r * 16 + half * 8 + v;
        const float val = acc[r][j][v] * scale + bv;
        if constexpr (OUT_BF16) {
          bf16_t* C = (bf16_t*)Cv + (size_t)z * sC;
          C[(size_t)row * ldc + col] = (bf16_t)val;
        } else {
          float* C = (float*)Cv + (size_t)z * sC;
          C[(size_t)row * ldc + col] = val;
        }
      }
    }
  }
}

// ---------------------------------------------------------------------------
// Row softmax over 1024 f32, writing bf16 result in place into the first half
// of each f32 row (row-private => race-free). One 256-thread block per row.
// ---------------------------------------------------------------------------
__global__ __launch_bounds__(256) void softmax_inplace_bf16(float* __restrict__ S) {
  const int row = blockIdx.x;
  float* rp = S + (size_t)row * 1024;
  const int tid = threadIdx.x;
  __shared__ float red[8];

  float v[4];
#pragma unroll
  for (int i = 0; i < 4; ++i) v[i] = rp[tid + i * 256];

  float m = fmaxf(fmaxf(v[0], v[1]), fmaxf(v[2], v[3]));
  for (int off = 16; off > 0; off >>= 1) m = fmaxf(m, __shfl_xor(m, off, 32));
  if ((tid & 31) == 0) red[tid >> 5] = m;
  __syncthreads();
  float gm = red[0];
#pragma unroll
  for (int j = 1; j < 8; ++j) gm = fmaxf(gm, red[j]);

  float s = 0.0f;
#pragma unroll
  for (int i = 0; i < 4; ++i) { v[i] = __expf(v[i] - gm); s += v[i]; }
  for (int off = 16; off > 0; off >>= 1) s += __shfl_xor(s, off, 32);
  __syncthreads();
  if ((tid & 31) == 0) red[tid >> 5] = s;
  __syncthreads();
  float gs = 0.0f;
#pragma unroll
  for (int j = 0; j < 8; ++j) gs += red[j];
  const float inv = 1.0f / gs;

  bf16_t* op = (bf16_t*)rp;                   // bf16 row, stride 2048 halfwords
#pragma unroll
  for (int i = 0; i < 4; ++i) op[tid + i * 256] = (bf16_t)(v[i] * inv);
}

// ---------------------------------------------------------------------------
// Host-side orchestration
// ---------------------------------------------------------------------------
extern "C" void kernel_launch(void* const* d_in, const int* in_sizes, int n_in,
                              void* d_out, int out_size, void* d_ws, size_t ws_size,
                              hipStream_t stream) {
  const int B = 16, T = 1024, H = 1024;
  const long long TH  = (long long)T * H;       // 1M
  const long long TT  = (long long)T * T;       // 1M (per-batch scores)

  const float* query = (const float*)d_in[0];
  const float* keys  = (const float*)d_in[1];
  const float* vals  = (const float*)d_in[2];
  const float* Wq    = (const float*)d_in[3];
  const float* bq    = (const float*)d_in[4];
  const float* Wk    = (const float*)d_in[5];
  const float* bk    = (const float*)d_in[6];
  const float* Wo    = (const float*)d_in[7];
  const float* bo    = (const float*)d_in[8];
  float* out = (float*)d_out;

  char* ws = (char*)d_ws;
  const size_t MB = 1024ull * 1024ull;
  bf16_t* qb  = (bf16_t*)(ws + 0 * MB);     // 32MB: q projection (bf16), reused for ctx
  bf16_t* kb  = (bf16_t*)(ws + 32 * MB);    // 32MB: k projection (bf16)
  bf16_t* vbT = (bf16_t*)(ws + 64 * MB);    // 32MB: V^T (bf16), [b][h][s]
  float*  S   = (float*)(ws + 96 * MB);     // 64MB: scores f32; attn bf16 in-place
  bf16_t* wqb = (bf16_t*)(ws + 160 * MB);   // 2MB
  bf16_t* wkb = (bf16_t*)(ws + 162 * MB);   // 2MB
  bf16_t* wob = (bf16_t*)(ws + 164 * MB);   // 2MB  (166MB workspace total)

  // 1) weight conversions + V transpose/convert (once; removes transpose work
  //    from the hot attn@V GEMM loop entirely)
  cvt_f32_bf16<<<1024, 256, 0, stream>>>(Wq, wqb, TH);
  cvt_f32_bf16<<<1024, 256, 0, stream>>>(Wk, wkb, TH);
  cvt_f32_bf16<<<1024, 256, 0, stream>>>(Wo, wob, TH);
  transpose_cvt_bf16<<<dim3(T / 32, H / 32, B), dim3(32, 8), 0, stream>>>(vals, vbT);

  dim3 blk(NT);

  // 2) q = query @ Wq^T + bq ; k = keys @ Wk^T + bk   (f32 A, bias, bf16 out)
  gemm_wmma<true, true, true><<<dim3(H / BN, (B * T) / BM, 1), blk, 0, stream>>>(
      query, wqb, bq, qb, B * T, H, H, H, H, H, 0, 0, 0, 1.0f);
  gemm_wmma<true, true, true><<<dim3(H / BN, (B * T) / BM, 1), blk, 0, stream>>>(
      keys, wkb, bk, kb, B * T, H, H, H, H, H, 0, 0, 0, 1.0f);

  // 3) scores: S[b] = qb[b] @ kb[b]^T  (bf16 A/B, f32 out), batched over z
  gemm_wmma<false, false, false><<<dim3(T / BN, T / BM, B), blk, 0, stream>>>(
      qb, kb, nullptr, S, T, T, H, H, H, T, TH, TH, TT, 1.0f);

  // 4) softmax over key axis; bf16 attn stored in place (row stride 2048 halfwords)
  softmax_inplace_bf16<<<B * T, 256, 0, stream>>>(S);

  // 5) ctx[b] = (attn[b] @ V[b]) / 32 -> bf16, reusing qb buffer.
  //    B operand = V^T (N=h rows, K=s contiguous) — fast row-major path.
  gemm_wmma<false, false, true><<<dim3(H / BN, T / BM, B), blk, 0, stream>>>(
      (const void*)S, vbT, nullptr, qb, T, H, T,
      2 * T, T, H, (long long)T * 2 * T, TH, TH, 1.0f / 32.0f);

  // 6) out = ctx @ Wo^T + bo  (bf16 A, bias, f32 out)
  gemm_wmma<false, true, false><<<dim3(H / BN, (B * T) / BM, 1), blk, 0, stream>>>(
      qb, wob, bo, out, B * T, H, H, H, H, H, 0, 0, 0, 1.0f);

  (void)in_sizes; (void)n_in; (void)out_size; (void)ws_size;
}
